// GAT_47906065219807
// MI455X (gfx1250) — compile-verified, matrix operand order checked by reading, not run on hardware
//
#include <hip/hip_runtime.h>
#include <hip/hip_bf16.h>

// ---------------------------------------------------------------------------
// GATv2 x3 + normalize + pool + MLP for MI455X (gfx1250, wave32, WMMA).
// GEMM inputs pre-converted to f16 in padded (Mp x Kp) buffers so the WMMA
// GEMM staging path is branchless b128 global->LDS copies. Double-buffered
// LDS: one barrier per K-step, ds_stores overlap WMMA issue. f32 accum.
// Edge softmax/aggregation via f32 global atomics (all hot buffers < L2).
// ---------------------------------------------------------------------------

typedef _Float16 v16h __attribute__((ext_vector_type(16)));
typedef float    v8f  __attribute__((ext_vector_type(8)));
typedef _Float16 h4_t __attribute__((ext_vector_type(4)));
typedef _Float16 h8_t __attribute__((ext_vector_type(8)));

#define N_NODES 20000
#define MP_NODES 20096      // 157 * 128, row-padded node count
#define E0_EDGES 120000
#define E_TOT   140000      // + self loops
#define N_GRAPHS 512
#define F_IN 300
#define KP1 320             // F_IN padded to multiple of 32
#define CCH 256             // channels per head
#define NOUT 768
#define SLOPE 0.2f
#define EPSA 1e-16f

#define BM 128
#define BN 128
#define BK 32
#define BKP 40              // padded LDS row stride in f16 (80B: conflict-free)
#define TILE_H (BM * BKP)   // f16 elements per LDS tile buffer

// ---------------- ordered-uint float max helpers ---------------------------
__device__ __forceinline__ unsigned f2ord(float f) {
    unsigned u = __float_as_uint(f);
    return (u & 0x80000000u) ? ~u : (u | 0x80000000u);
}
__device__ __forceinline__ float ord2f(unsigned u) {
    return (u & 0x80000000u) ? __uint_as_float(u & 0x7FFFFFFFu)
                             : __uint_as_float(~u);
}
#define NEG_INF_ORD 0x007FFFFFu   // f2ord(-inf)

// ---------------- vectorized fill ------------------------------------------
__global__ void fill_u32x4(uint4* __restrict__ p, unsigned v, int n4) {
    int i = blockIdx.x * blockDim.x + threadIdx.x;
    if (i < n4) p[i] = make_uint4(v, v, v, v);
}

// ---------------- f32 -> padded f16 convert --------------------------------
// Xh is [Mp x Kp] f16, zero-filled outside [M x K]. K and Kp multiples of 4.
__global__ void cvt_f16(const float* __restrict__ X, int M, int K,
                        _Float16* __restrict__ Xh, int Kp, int n4)
{
    int g = blockIdx.x * blockDim.x + threadIdx.x;
    if (g >= n4) return;
    int kp4 = Kp >> 2;
    int row = g / kp4;
    int col = (g - row * kp4) * 4;
    h4_t h;
    if (row < M && col < K) {
        float4 v = *(const float4*)(X + (size_t)row * K + col);
        h[0] = (_Float16)v.x; h[1] = (_Float16)v.y;
        h[2] = (_Float16)v.z; h[3] = (_Float16)v.w;
    } else {
        h[0] = h[1] = h[2] = h[3] = (_Float16)0.f;
    }
    *(h4_t*)(Xh + (size_t)row * Kp + col) = h;
}

// ---------------- WMMA GEMM: C[M,N] = Xh[M,Kp] * Wh[N,Kp]^T ----------------
// Xh: [gridDim.x*128 x Kp] f16 (zero-padded rows). Wh: [N x Kp] f16.
// Block: 256 threads (8 waves). Block tile 128x128, K-step 32.
// Wave w: rows (w/2)*32, cols (w&1)*64 -> 2x4 wmma accumulators.
// Double-buffered LDS (writes go to the opposite buffer of reads) -> a
// single barrier per K-step; register-staged b128 prefetch of tile ks+2.
__global__ __launch_bounds__(256) void gemm_nt_f16(
    const _Float16* __restrict__ Xh, const _Float16* __restrict__ Wh,
    float* __restrict__ C, int M, int N, int Kp,
    const float* __restrict__ bias, int act)
{
    __shared__ __align__(16) _Float16 As[2 * TILE_H];
    __shared__ __align__(16) _Float16 Ws[2 * TILE_H];

    const int bm   = blockIdx.x * BM;
    const int bn   = blockIdx.y * BN;
    const int tid  = threadIdx.x;
    const int wave = tid >> 5;
    const int lane = tid & 31;
    const int wm   = (wave >> 1) * 32;
    const int wn   = (wave & 1) * 64;
    const int half = lane >> 4;
    const int lr   = lane & 15;

    v8f acc[2][4] = {};
    h8_t ar[2], wreg[2];

    // per-thread staging geometry: 512 f16x8 groups per 128x32 tile
    const int row0 = tid >> 2,         kof0 = (tid & 3) * 8;
    const int row1 = (tid + 256) >> 2, kof1 = ((tid + 256) & 3) * 8;

    const int ksteps = Kp / BK;

    auto ldreg = [&](int k0) {
        ar[0]   = *(const h8_t*)(Xh + (size_t)(bm + row0) * Kp + k0 + kof0);
        ar[1]   = *(const h8_t*)(Xh + (size_t)(bm + row1) * Kp + k0 + kof1);
        wreg[0] = *(const h8_t*)(Wh + (size_t)(bn + row0) * Kp + k0 + kof0);
        wreg[1] = *(const h8_t*)(Wh + (size_t)(bn + row1) * Kp + k0 + kof1);
    };
    auto stlds = [&](int buf) {
        _Float16* a = As + buf * TILE_H;
        _Float16* w = Ws + buf * TILE_H;
        *(h8_t*)&a[row0 * BKP + kof0] = ar[0];
        *(h8_t*)&a[row1 * BKP + kof1] = ar[1];
        *(h8_t*)&w[row0 * BKP + kof0] = wreg[0];
        *(h8_t*)&w[row1 * BKP + kof1] = wreg[1];
    };

    // prologue: tile 0 -> LDS buf 0; tile 1 -> registers
    ldreg(0);
    stlds(0);
    if (ksteps > 1) ldreg(BK);
    __syncthreads();

    for (int ks = 0; ks < ksteps; ++ks) {
        const int cur = ks & 1;
        const _Float16* Ab = As + cur * TILE_H;
        const _Float16* Wb = Ws + cur * TILE_H;

        // fragment gather per the 16-bit 16x32 layout:
        // vgpr pair j: k = (j>>2)*16 + half*8 + (j&3)*2
        v16h afrag[2], bfrag[4];
        #pragma unroll
        for (int mi = 0; mi < 2; ++mi) {
            const _Float16* ap = &Ab[(wm + mi * 16 + lr) * BKP];
            #pragma unroll
            for (int j = 0; j < 8; ++j) {
                int k = (j >> 2) * 16 + half * 8 + (j & 3) * 2;
                afrag[mi][2 * j]     = ap[k];
                afrag[mi][2 * j + 1] = ap[k + 1];
            }
        }
        #pragma unroll
        for (int ni = 0; ni < 4; ++ni) {
            const _Float16* bp = &Wb[(wn + ni * 16 + lr) * BKP];
            #pragma unroll
            for (int j = 0; j < 8; ++j) {
                int k = (j >> 2) * 16 + half * 8 + (j & 3) * 2;
                bfrag[ni][2 * j]     = bp[k];
                bfrag[ni][2 * j + 1] = bp[k + 1];
            }
        }

        // stage tile ks+1 into the opposite LDS buffer (overlaps WMMAs),
        // then prefetch tile ks+2 into registers.
        if (ks + 1 < ksteps) {
            stlds((ks + 1) & 1);
            if (ks + 2 < ksteps) ldreg((ks + 2) * BK);
        }

        #pragma unroll
        for (int mi = 0; mi < 2; ++mi)
            #pragma unroll
            for (int ni = 0; ni < 4; ++ni)
                acc[mi][ni] = __builtin_amdgcn_wmma_f32_16x16x32_f16(
                    false, afrag[mi], false, bfrag[ni],
                    (short)0, acc[mi][ni], false, false);

        // one barrier per K-step: (a) all reads of buf[cur] done before it is
        // overwritten next step, (b) buf[(ks+1)&1] stores visible next step.
        __syncthreads();
    }

    // epilogue: D layout -> vgpr r, lane: m = r + half*8, n = lane&15
    #pragma unroll
    for (int mi = 0; mi < 2; ++mi) {
        #pragma unroll
        for (int ni = 0; ni < 4; ++ni) {
            #pragma unroll
            for (int r = 0; r < 8; ++r) {
                int row = bm + wm + mi * 16 + r + half * 8;
                int col = bn + wn + ni * 16 + lr;
                if (row < M && col < N) {
                    float v = acc[mi][ni][r];
                    if (bias) v += bias[col];
                    if (act == 1) v = v > 0.0f ? v : 0.0f;
                    C[(size_t)row * N + col] = v;
                }
            }
        }
    }
}

// ---------------- edge helpers ---------------------------------------------
__device__ __forceinline__ void edge_sd(const int* __restrict__ ei, int e,
                                        int& s, int& d) {
    if (e < E0_EDGES) { s = ei[e]; d = ei[E0_EDGES + e]; }
    else              { s = e - E0_EDGES; d = s; }
}

// one wave per (edge, head): score = att_h . leaky_relu(xl[src]+xr[dst])
__global__ __launch_bounds__(256) void edge_scores(
    const float* __restrict__ xl, const float* __restrict__ xr,
    const float* __restrict__ att, const int* __restrict__ ei,
    int H, float* __restrict__ scores)
{
    int gw   = blockIdx.x * 8 + (threadIdx.x >> 5);
    int lane = threadIdx.x & 31;
    if (gw >= E_TOT * H) return;
    int e = gw / H, h = gw - e * H;
    int s, d; edge_sd(ei, e, s, d);
    const int HC = H * CCH;
    const float* pl = xl + (size_t)s * HC + h * CCH;
    const float* pr = xr + (size_t)d * HC + h * CCH;
    const float* pa = att + h * CCH;
    float acc = 0.0f;
    #pragma unroll
    for (int t = 0; t < 2; ++t) {
        int c = (t * 32 + lane) * 4;
        float4 a = *(const float4*)(pl + c);
        float4 b = *(const float4*)(pr + c);
        float4 w = *(const float4*)(pa + c);
        float v0 = a.x + b.x, v1 = a.y + b.y, v2 = a.z + b.z, v3 = a.w + b.w;
        v0 = v0 > 0.f ? v0 : SLOPE * v0;
        v1 = v1 > 0.f ? v1 : SLOPE * v1;
        v2 = v2 > 0.f ? v2 : SLOPE * v2;
        v3 = v3 > 0.f ? v3 : SLOPE * v3;
        acc += v0 * w.x + v1 * w.y + v2 * w.z + v3 * w.w;
    }
    #pragma unroll
    for (int m = 16; m; m >>= 1) acc += __shfl_xor(acc, m, 32);
    if (lane == 0) scores[(size_t)e * H + h] = acc;
}

__global__ void edge_segmax(const float* __restrict__ scores,
                            const int* __restrict__ ei, int H,
                            unsigned* __restrict__ mx)
{
    int i = blockIdx.x * blockDim.x + threadIdx.x;
    if (i >= E_TOT * H) return;
    int e = i / H, h = i - e * H;
    int s, d; edge_sd(ei, e, s, d); (void)s;
    atomicMax(&mx[(size_t)d * H + h], f2ord(scores[i]));
}

__global__ void edge_expdenom(float* __restrict__ scores,
                              const int* __restrict__ ei, int H,
                              const unsigned* __restrict__ mx,
                              float* __restrict__ denom)
{
    int i = blockIdx.x * blockDim.x + threadIdx.x;
    if (i >= E_TOT * H) return;
    int e = i / H, h = i - e * H;
    int s, d; edge_sd(ei, e, s, d); (void)s;
    float m  = ord2f(mx[(size_t)d * H + h]);
    float ex = expf(scores[i] - m);
    scores[i] = ex;
    atomicAdd(&denom[(size_t)d * H + h], ex);
}

// one wave per (edge, head): out[dst,h,:] += alpha * xl[src,h,:]
__global__ __launch_bounds__(256) void edge_aggregate(
    const float* __restrict__ ex, const float* __restrict__ denom,
    const float* __restrict__ xl, const int* __restrict__ ei, int H,
    float* __restrict__ out)
{
    int gw   = blockIdx.x * 8 + (threadIdx.x >> 5);
    int lane = threadIdx.x & 31;
    if (gw >= E_TOT * H) return;
    int e = gw / H, h = gw - e * H;
    int s, d; edge_sd(ei, e, s, d);
    float alpha = ex[(size_t)e * H + h] / (denom[(size_t)d * H + h] + EPSA);
    const int HC = H * CCH;
    const float* pl = xl + (size_t)s * HC + h * CCH;
    float*       po = out + (size_t)d * HC + h * CCH;
    #pragma unroll
    for (int t = 0; t < 2; ++t) {
        int c = (t * 32 + lane) * 4;
        float4 a = *(const float4*)(pl + c);
        atomicAdd(&po[c + 0], alpha * a.x);
        atomicAdd(&po[c + 1], alpha * a.y);
        atomicAdd(&po[c + 2], alpha * a.z);
        atomicAdd(&po[c + 3], alpha * a.w);
    }
}

// h = elu(agg + bias) written as f16 directly into the padded GEMM input
// buffer (row stride == HC here, so the layout is linear). 4 elems/thread.
__global__ void bias_elu_f16(const float* __restrict__ in,
                             const float* __restrict__ b, int HC,
                             _Float16* __restrict__ out, int n4)
{
    int i = blockIdx.x * blockDim.x + threadIdx.x;
    if (i >= n4) return;
    int c = i * 4;
    int bb = c % HC;
    float4 v = *(const float4*)(in + c);
    h4_t o;
    float t;
    t = v.x + b[bb + 0]; o[0] = (_Float16)(t > 0.f ? t : (expf(t) - 1.f));
    t = v.y + b[bb + 1]; o[1] = (_Float16)(t > 0.f ? t : (expf(t) - 1.f));
    t = v.z + b[bb + 2]; o[2] = (_Float16)(t > 0.f ? t : (expf(t) - 1.f));
    t = v.w + b[bb + 3]; o[3] = (_Float16)(t > 0.f ? t : (expf(t) - 1.f));
    *(h4_t*)(out + c) = o;
}

// conv3: mean over 6 heads + bias -> [N,256]; 4 channels per thread
__global__ void mean_bias6(const float* __restrict__ agg,
                           const float* __restrict__ b,
                           float* __restrict__ out)
{
    int i = blockIdx.x * blockDim.x + threadIdx.x;   // float4 index
    if (i >= N_NODES * CCH / 4) return;
    int node = i >> 6;           // 64 groups per node
    int c    = (i & 63) * 4;
    const float* p = agg + (size_t)node * (6 * CCH) + c;
    float4 s = make_float4(0.f, 0.f, 0.f, 0.f);
    #pragma unroll
    for (int h = 0; h < 6; ++h) {
        float4 v = *(const float4*)(p + h * CCH);
        s.x += v.x; s.y += v.y; s.z += v.z; s.w += v.w;
    }
    const float inv6 = 1.0f / 6.0f;
    float4 o;
    o.x = s.x * inv6 + b[c + 0];
    o.y = s.y * inv6 + b[c + 1];
    o.z = s.z * inv6 + b[c + 2];
    o.w = s.w * inv6 + b[c + 3];
    *(float4*)(out + (size_t)node * CCH + c) = o;
}

// one wave per node: L2 normalize row of 256, then atomically mean-pool
__global__ __launch_bounds__(256) void normalize_pool(
    float* __restrict__ hf, const int* __restrict__ batch,
    float* __restrict__ pool, float* __restrict__ cnt)
{
    int node = blockIdx.x * 8 + (threadIdx.x >> 5);
    int lane = threadIdx.x & 31;
    if (node >= N_NODES) return;
    float* p = hf + (size_t)node * CCH;
    float4 vals[2];
    float ss = 0.0f;
    #pragma unroll
    for (int t = 0; t < 2; ++t) {
        int c = (t * 32 + lane) * 4;
        vals[t] = *(const float4*)(p + c);
        ss += vals[t].x * vals[t].x + vals[t].y * vals[t].y
            + vals[t].z * vals[t].z + vals[t].w * vals[t].w;
    }
    #pragma unroll
    for (int m = 16; m; m >>= 1) ss += __shfl_xor(ss, m, 32);
    float inv = 1.0f / fmaxf(sqrtf(ss), 1e-12f);
    int g = batch[node];
    float* pg = pool + (size_t)g * CCH;
    #pragma unroll
    for (int t = 0; t < 2; ++t) {
        int c = (t * 32 + lane) * 4;
        float4 v;
        v.x = vals[t].x * inv; v.y = vals[t].y * inv;
        v.z = vals[t].z * inv; v.w = vals[t].w * inv;
        *(float4*)(p + c) = v;
        atomicAdd(&pg[c + 0], v.x);
        atomicAdd(&pg[c + 1], v.y);
        atomicAdd(&pg[c + 2], v.z);
        atomicAdd(&pg[c + 3], v.w);
    }
    if (lane == 0) atomicAdd(&cnt[g], 1.0f);
}

__global__ void pool_div(float* __restrict__ pool,
                         const float* __restrict__ cnt)
{
    int i = blockIdx.x * blockDim.x + threadIdx.x;
    if (i >= N_GRAPHS * CCH) return;
    pool[i] /= fmaxf(cnt[i >> 8], 1.0f);
}

// ---------------------------------------------------------------------------
extern "C" void kernel_launch(void* const* d_in, const int* in_sizes, int n_in,
                              void* d_out, int out_size, void* d_ws, size_t ws_size,
                              hipStream_t stream)
{
    const float* x      = (const float*)d_in[0];
    const int*   ei     = (const int*)  d_in[1];
    const int*   batch  = (const int*)  d_in[2];
    const float* c1_wl  = (const float*)d_in[3];
    const float* c1_wr  = (const float*)d_in[4];
    const float* c1_att = (const float*)d_in[5];
    const float* c1_b   = (const float*)d_in[6];
    const float* c2_wl  = (const float*)d_in[7];
    const float* c2_wr  = (const float*)d_in[8];
    const float* c2_att = (const float*)d_in[9];
    const float* c2_b   = (const float*)d_in[10];
    const float* c3_wl  = (const float*)d_in[11];
    const float* c3_wr  = (const float*)d_in[12];
    const float* c3_att = (const float*)d_in[13];
    const float* c3_b   = (const float*)d_in[14];
    const float* fp1_w  = (const float*)d_in[15];
    const float* fp1_b  = (const float*)d_in[16];
    const float* fp2_w  = (const float*)d_in[17];
    const float* fp2_b  = (const float*)d_in[18];

    // -------- workspace carve (256B aligned) ------------------------------
    char* base = (char*)d_ws;
    size_t off = 0;
    auto carve = [&](size_t bytes) -> char* {
        char* p = base + off;
        off += (bytes + 255) & ~(size_t)255;
        return p;
    };
    const size_t NB = (size_t)N_NODES * 1536 * sizeof(float); // widest layer
    float*     bufA = (float*)carve(NB);                      // xl (f32)
    float*     bufB = (float*)carve(NB);                      // xr -> agg (f32)
    _Float16*  Xh   = (_Float16*)carve((size_t)MP_NODES * 1024 * sizeof(_Float16));
    _Float16*  Whl  = (_Float16*)carve((size_t)1536 * 1024 * sizeof(_Float16));
    _Float16*  Whr  = (_Float16*)carve((size_t)1536 * 1024 * sizeof(_Float16));
    float* scores = (float*)carve((size_t)E_TOT * 6 * sizeof(float));
    unsigned* mx  = (unsigned*)carve((size_t)N_NODES * 6 * sizeof(unsigned));
    float* denom  = (float*)carve((size_t)N_NODES * 6 * sizeof(float));
    float* hf     = (float*)carve((size_t)N_NODES * CCH * sizeof(float));
    float* pool   = (float*)carve((size_t)N_GRAPHS * CCH * sizeof(float));
    float* cnt    = (float*)carve((size_t)N_GRAPHS * sizeof(float));
    float* mlp1   = (float*)carve((size_t)N_GRAPHS * CCH * sizeof(float));
    (void)ws_size; (void)n_in; (void)in_sizes; (void)out_size;

    auto fill4 = [&](void* p, unsigned v, int n_u32) {
        int n4 = n_u32 / 4;   // all our sizes are multiples of 4 u32
        fill_u32x4<<<(n4 + 255) / 256, 256, 0, stream>>>((uint4*)p, v, n4);
    };
    auto cvt = [&](const float* src, int M, int K, _Float16* dst, int Mp, int Kp) {
        int n4 = Mp * Kp / 4;
        cvt_f16<<<(n4 + 255) / 256, 256, 0, stream>>>(src, M, K, dst, Kp, n4);
    };

    // zero the padded tail rows of Xh for the stride-1024 layers (conv2/3
    // inputs written by bias_elu_f16, which only covers rows < N_NODES).
    fill4(Xh + (size_t)N_NODES * 1024, 0u, (MP_NODES - N_NODES) * 1024 / 2);

    // attention block (GEMM inputs Xh/Whl/Whr already converted & padded)
    auto run_conv = [&](int Kp, int H, const float* att) {
        const int HC = H * CCH;
        dim3 gg(MP_NODES / BM, HC / BN);
        gemm_nt_f16<<<gg, 256, 0, stream>>>(Xh, Whl, bufA, N_NODES, HC, Kp, nullptr, 0);
        gemm_nt_f16<<<gg, 256, 0, stream>>>(Xh, Whr, bufB, N_NODES, HC, Kp, nullptr, 0);
        const int tot = E_TOT * H;
        edge_scores<<<(tot + 7) / 8, 256, 0, stream>>>(bufA, bufB, att, ei, H, scores);
        fill4(mx, NEG_INF_ORD, N_NODES * H);
        edge_segmax<<<(tot + 255) / 256, 256, 0, stream>>>(scores, ei, H, mx);
        fill4(denom, 0u, N_NODES * H);
        edge_expdenom<<<(tot + 255) / 256, 256, 0, stream>>>(scores, ei, H, mx, denom);
        // xr dead after scoring -> reuse bufB as aggregation target
        fill4(bufB, 0u, N_NODES * HC);
        edge_aggregate<<<(tot + 7) / 8, 256, 0, stream>>>(scores, denom, bufA, ei, H, bufB);
    };

    // ---- conv1: x [20000,300] -> Xh [20096,320]; H=4 ----------------------
    cvt(x, N_NODES, F_IN, Xh, MP_NODES, KP1);
    cvt(c1_wl, 1024, F_IN, Whl, 1024, KP1);
    cvt(c1_wr, 1024, F_IN, Whr, 1024, KP1);
    run_conv(KP1, 4, c1_att);
    bias_elu_f16<<<(N_NODES * 1024 / 4 + 255) / 256, 256, 0, stream>>>(
        bufB, c1_b, 1024, Xh, N_NODES * 1024 / 4);   // h1 -> Xh [.,1024]

    // ---- conv2: Xh [20096,1024]; H=4 --------------------------------------
    cvt(c2_wl, 1024, 1024, Whl, 1024, 1024);
    cvt(c2_wr, 1024, 1024, Whr, 1024, 1024);
    run_conv(1024, 4, c2_att);
    bias_elu_f16<<<(N_NODES * 1024 / 4 + 255) / 256, 256, 0, stream>>>(
        bufB, c2_b, 1024, Xh, N_NODES * 1024 / 4);   // h2 -> Xh [.,1024]

    // ---- conv3: Xh [20096,1024]; H=6, mean over heads ---------------------
    cvt(c3_wl, 1536, 1024, Whl, 1536, 1024);
    cvt(c3_wr, 1536, 1024, Whr, 1536, 1024);
    run_conv(1024, 6, c3_att);
    mean_bias6<<<(N_NODES * CCH / 4 + 255) / 256, 256, 0, stream>>>(bufB, c3_b, hf);

    // ---- L2 normalize + global mean pool ----------------------------------
    fill4(pool, 0u, N_GRAPHS * CCH);
    fill4(cnt, 0u, N_GRAPHS);
    normalize_pool<<<(N_NODES + 7) / 8, 256, 0, stream>>>(hf, batch, pool, cnt);
    pool_div<<<(N_GRAPHS * CCH + 255) / 256, 256, 0, stream>>>(pool, cnt);

    // ---- MLP: relu(pool @ fp1_w^T + b1) @ fp2_w^T + b2 -> d_out [512,768] -
    cvt(pool, N_GRAPHS, CCH, Xh, N_GRAPHS, CCH);
    cvt(fp1_w, CCH, CCH, Whl, CCH, CCH);
    dim3 g1(N_GRAPHS / BM, CCH / BN);
    gemm_nt_f16<<<g1, 256, 0, stream>>>(Xh, Whl, mlp1, N_GRAPHS, CCH, CCH, fp1_b, 1);

    cvt(mlp1, N_GRAPHS, CCH, Xh, N_GRAPHS, CCH);
    cvt(fp2_w, NOUT, CCH, Whl, NOUT, CCH);
    dim3 g2(N_GRAPHS / BM, NOUT / BN);
    gemm_nt_f16<<<g2, 256, 0, stream>>>(Xh, Whl, (float*)d_out, N_GRAPHS, NOUT, CCH, fp2_b, 0);
}
